// DAGAwareAttention_63084479643821
// MI455X (gfx1250) — compile-verified, matrix operand order checked by reading, not run on hardware
//
#include <hip/hip_runtime.h>

typedef __attribute__((ext_vector_type(16))) _Float16 v16h;
typedef __attribute__((ext_vector_type(8)))  _Float16 v8h;
typedef __attribute__((ext_vector_type(8)))  float    v8f;
typedef int v4i_vs __attribute__((vector_size(16))); // matches builtin param

#define WMMA_F16(a, b, c) \
    __builtin_amdgcn_wmma_f32_16x16x32_f16(false, (a), false, (b), (short)0, (c), false, false)

static constexpr int   kB  = 2;
static constexpr int   kS  = 4096;
static constexpr int   kDM = 512;
static constexpr int   kH  = 8;
static constexpr int   kDK = 64;
static constexpr float kScale = 0.125f; // 1/sqrt(64)

#if __has_builtin(__builtin_amdgcn_global_load_async_to_lds_b128)
#define ASYNC_LDS 1
#endif

// 16-byte global -> LDS copy. Async DMA path on gfx1250
// (GLOBAL_LOAD_ASYNC_TO_LDS_B128, ASYNCcnt), else load+ds_store fallback.
__device__ __forceinline__ void cp16_g2s(const _Float16* g, _Float16* l) {
#ifdef ASYNC_LDS
    __builtin_amdgcn_global_load_async_to_lds_b128(
        (__attribute__((address_space(1))) v4i_vs*)g,
        (__attribute__((address_space(3))) v4i_vs*)l, 0, 0);
#else
    *(v8h*)l = *(const v8h*)g;
#endif
}

__device__ __forceinline__ void wait_g2s() {
#ifdef ASYNC_LDS
#if __has_builtin(__builtin_amdgcn_s_wait_asynccnt)
    __builtin_amdgcn_s_wait_asynccnt(0);
#else
    asm volatile("s_wait_asynccnt 0" ::: "memory");
#endif
#endif
}

// ---------------------------------------------------------------------------
// Kernel 0: pack the (S,S) int32 DAG mask into 1 bit/entry.
// bits[q][c] bit j == (mask[q][c*32 + j] != 0). 2MB result; read 16x by the
// attention kernel instead of 16x 67MB of int32 -> 32x less mask traffic.
// ---------------------------------------------------------------------------
__global__ __launch_bounds__(256) void k_mask_pack(
    const int* __restrict__ mask, unsigned* __restrict__ bits)
{
    const size_t w = (size_t)blockIdx.x * 256 + threadIdx.x; // word index
    const int4* src = (const int4*)(mask + w * 32);
    unsigned word = 0;
#pragma unroll
    for (int j = 0; j < 8; ++j) {
        const int4 m = src[j];
        word |= (m.x ? 1u : 0u) << (4 * j + 0);
        word |= (m.y ? 1u : 0u) << (4 * j + 1);
        word |= (m.z ? 1u : 0u) << (4 * j + 2);
        word |= (m.w ? 1u : 0u) << (4 * j + 3);
    }
    bits[w] = word;
}

// ---------------------------------------------------------------------------
// Kernel 1: fused QKV projection (all three in one wave).
//   Y[m][n] = sum_k X[m][k] * W[n][k] + bias[n]   (einsum 'bsd,ed->bse')
// One wave computes a 16(m) x 64(n) tile for Q, K AND V; the converted
// x A-tile is reused across the three weight matrices (12 WMMAs / A-load).
// Q,K stored f16 as (B,H,S,DK); V stored f16 TRANSPOSED as (B,H,DK,S).
//
// WMMA operand convention (CDNA5 wave32, 16x16x32 f16):
//   A (MxK): lanes 0-15 hold row M=lane, K elems {0..7,16..23};
//            lanes 16-31 hold row M=lane-16, K elems {8..15,24..31}.
//   B (KxN): lane n<16 holds col N=n with K elems {0..7,16..23};
//            lane n+16 holds col N=n with K elems {8..15,24..31}.
//   C/D    : VGPR r, lanes 0-15 -> (M=r, N=lane); lanes 16-31 -> (M=8+r).
// ---------------------------------------------------------------------------
__global__ __launch_bounds__(32) void k_proj_qkv(
    const float* __restrict__ x,
    const float* __restrict__ wq, const float* __restrict__ bq,
    const float* __restrict__ wk, const float* __restrict__ bk,
    const float* __restrict__ wv, const float* __restrict__ bv,
    _Float16* __restrict__ qf, _Float16* __restrict__ kf,
    _Float16* __restrict__ vf)
{
    const int lane = threadIdx.x & 31;
    const int half = lane >> 4;
    const int lm   = lane & 15;
    const int koff = half ? 8 : 0;

    const int m0 = blockIdx.x * 16;   // row tile over B*S = 8192 rows
    const int h  = blockIdx.y;        // head -> n0 = h*64
    const int n0 = h * kDK;

    const float* wmat[3] = {wq, wk, wv};
    const float* bias[3] = {bq, bk, bv};

    v8f acc[3][4] = {};

    const float* xr = x + (size_t)(m0 + lm) * kDM;
    for (int k0 = 0; k0 < kDM; k0 += 32) {
        v16h a;
#pragma unroll
        for (int i = 0; i < 8; ++i) {
            a[i]     = (_Float16)xr[k0 + koff + i];
            a[i + 8] = (_Float16)xr[k0 + 16 + koff + i];
        }
#pragma unroll
        for (int wsel = 0; wsel < 3; ++wsel) {
#pragma unroll
            for (int t = 0; t < 4; ++t) {
                const float* wr =
                    wmat[wsel] + (size_t)(n0 + t * 16 + lm) * kDM + k0;
                v16h bm;
#pragma unroll
                for (int i = 0; i < 8; ++i) {
                    bm[i]     = (_Float16)wr[koff + i];
                    bm[i + 8] = (_Float16)wr[16 + koff + i];
                }
                acc[wsel][t] = WMMA_F16(a, bm, acc[wsel][t]);
            }
        }
    }

#pragma unroll
    for (int wsel = 0; wsel < 3; ++wsel) {
#pragma unroll
        for (int t = 0; t < 4; ++t) {
            const int dk = t * 16 + lm;          // N index inside head
            const float bval = bias[wsel][n0 + dk];
#pragma unroll
            for (int r = 0; r < 8; ++r) {
                const int m = m0 + r + half * 8; // global row (b*S + s)
                const int b = m >> 12;           // / 4096
                const int s = m & (kS - 1);
                const _Float16 val = (_Float16)(acc[wsel][t][r] + bval);
                if (wsel == 2) {                 // V transposed: (B,H,DK,S)
                    vf[(((size_t)b * kH + h) * kDK + dk) * kS + s] = val;
                } else {                         // Q/K: (B,H,S,DK)
                    _Float16* dst = (wsel == 0) ? qf : kf;
                    dst[(((size_t)b * kH + h) * kS + s) * kDK + dk] = val;
                }
            }
        }
    }
}

// ---------------------------------------------------------------------------
// Kernel 2: flash attention, transposed-score formulation, LDS-staged KV.
// Block = 256 threads = 8 waves, all on ONE (b,h); wave w owns q-tile
// (qgroup*8 + w). Each 32-kv chunk of K (32x64 f16) and V^T (64x32 f16) is
// staged into LDS once per block via async-to-LDS (double buffered), then
// all 8 waves consume it with ds_load_b128 -> 8x global traffic reduction.
// Mask comes from the packed bitmask: ONE u32 per lane per chunk.
//
//   S^T (kv x q): c0 (kv 0..15), c1 (kv 16..31), each = 2 WMMAs over DK=64.
//   O^T (dk x q): 4 tiles, A = V^T(dk x kv32), B = P^T (lane-local repack).
// Softmax rows (fixed q) live on the lane axis -> one shfl_xor(16) per
// reduction. Output written f16 to (B,S,H*DK).
// ---------------------------------------------------------------------------
__global__ __launch_bounds__(256) void k_attn(
    const _Float16* __restrict__ qf, const _Float16* __restrict__ kf,
    const _Float16* __restrict__ vf, const unsigned* __restrict__ mbits,
    _Float16* __restrict__ of)
{
    __shared__ alignas(16) _Float16 sK[2][32 * kDK]; // [kv][dk], 4KB/buf
    __shared__ alignas(16) _Float16 sV[2][kDK * 32]; // [dk][kv], 4KB/buf

    const int tid   = threadIdx.x;
    const int wid   = tid >> 5;
    const int lane  = tid & 31;
    const int half  = lane >> 4;
    const int lm    = lane & 15;
    const int koff  = half ? 8 : 0;

    const int qgroup = blockIdx.x & 31;         // 32 q-groups per (b,h)
    const int h      = (blockIdx.x >> 5) & (kH - 1);
    const int b      = blockIdx.x >> 8;
    const int q0     = (qgroup * 8 + wid) * 16;

    const _Float16* Qb = qf + (((size_t)b * kH + h) * kS) * kDK;
    const _Float16* Kb = kf + (((size_t)b * kH + h) * kS) * kDK;
    const _Float16* Vb = vf + (((size_t)b * kH + h) * kDK) * kS;

    // Per-thread staging slots: K chunk = 32 rows x 8 segs, V = 64 rows x 4.
    const int krow = tid >> 3, kseg = tid & 7;
    const int vrow = tid >> 2, vseg = tid & 3;

    // B operand of the score WMMAs: Q^T columns (fixed for the whole tile).
    v16h bq0, bq1;
    {
        const _Float16* qr = Qb + (size_t)(q0 + lm) * kDK;
        const v8h q_lo0 = *(const v8h*)(qr + koff);
        const v8h q_hi0 = *(const v8h*)(qr + 16 + koff);
        const v8h q_lo1 = *(const v8h*)(qr + 32 + koff);
        const v8h q_hi1 = *(const v8h*)(qr + 48 + koff);
#pragma unroll
        for (int i = 0; i < 8; ++i) {
            bq0[i] = q_lo0[i]; bq0[i + 8] = q_hi0[i];
            bq1[i] = q_lo1[i]; bq1[i + 8] = q_hi1[i];
        }
    }

    float mrow = -1e30f;
    float lrow = 0.0f;
    v8f o0 = {}, o1 = {}, o2 = {}, o3 = {};

    const int q = q0 + lm;
    const unsigned* mrowbits = mbits + (size_t)q * (kS / 32);

    // Prologue: stage chunk 0 into buffer 0.
    cp16_g2s(Kb + (size_t)krow * kDK + kseg * 8, &sK[0][krow * kDK + kseg * 8]);
    cp16_g2s(Vb + (size_t)vrow * kS + vseg * 8,  &sV[0][vrow * 32  + vseg * 8]);

    const int nchunks = kS / 32; // 128
    for (int ci = 0; ci < nchunks; ++ci) {
        const int cur = ci & 1;
        const int kv0 = ci * 32;

        wait_g2s();
        __syncthreads(); // chunk ci visible in LDS to all waves

        if (ci + 1 < nchunks) { // overlap next chunk's DMA with compute
            const int nkv = kv0 + 32;
            cp16_g2s(Kb + (size_t)(nkv + krow) * kDK + kseg * 8,
                     &sK[1 - cur][krow * kDK + kseg * 8]);
            cp16_g2s(Vb + (size_t)vrow * kS + nkv + vseg * 8,
                     &sV[1 - cur][vrow * 32 + vseg * 8]);
            __builtin_prefetch(mrowbits + ci + 1, 0, 1); // global_prefetch_b8
        }

        // ---- S^T tiles from LDS K rows
        v8f c0 = {}, c1 = {};
        {
            const _Float16* kr = &sK[cur][lm * kDK];
            v16h a;
            v8h lo = *(const v8h*)(kr + koff);
            v8h hi = *(const v8h*)(kr + 16 + koff);
#pragma unroll
            for (int i = 0; i < 8; ++i) { a[i] = lo[i]; a[i + 8] = hi[i]; }
            c0 = WMMA_F16(a, bq0, c0);
            lo = *(const v8h*)(kr + 32 + koff);
            hi = *(const v8h*)(kr + 48 + koff);
#pragma unroll
            for (int i = 0; i < 8; ++i) { a[i] = lo[i]; a[i + 8] = hi[i]; }
            c0 = WMMA_F16(a, bq1, c0);

            kr = &sK[cur][(16 + lm) * kDK];
            lo = *(const v8h*)(kr + koff);
            hi = *(const v8h*)(kr + 16 + koff);
#pragma unroll
            for (int i = 0; i < 8; ++i) { a[i] = lo[i]; a[i + 8] = hi[i]; }
            c1 = WMMA_F16(a, bq0, c1);
            lo = *(const v8h*)(kr + 32 + koff);
            hi = *(const v8h*)(kr + 48 + koff);
#pragma unroll
            for (int i = 0; i < 8; ++i) { a[i] = lo[i]; a[i + 8] = hi[i]; }
            c1 = WMMA_F16(a, bq1, c1);
        }

        // ---- mask (1 u32 for 32 kv) + online softmax
        const unsigned mw = mrowbits[ci]; // bit j = mask[q][kv0+j]
        float s0[8], s1[8];
        float tmax = -1e30f;
#pragma unroll
        for (int r = 0; r < 8; ++r) {
            const bool b0 = (mw >> (koff + r)) & 1u;
            const bool b1 = (mw >> (16 + koff + r)) & 1u;
            s0[r] = b0 ? c0[r] * kScale : -1e30f;
            s1[r] = b1 ? c1[r] * kScale : -1e30f;
            tmax = fmaxf(tmax, fmaxf(s0[r], s1[r]));
        }
        tmax = fmaxf(tmax, __shfl_xor(tmax, 16, 32));
        const float mnew  = fmaxf(mrow, tmax);
        const float alpha = __expf(mrow - mnew);

        float psum = 0.0f;
        v16h bp; // P^T repacked lane-locally into the B-operand layout
#pragma unroll
        for (int r = 0; r < 8; ++r) {
            const bool b0 = (mw >> (koff + r)) & 1u;
            const bool b1 = (mw >> (16 + koff + r)) & 1u;
            const float p0 = b0 ? __expf(s0[r] - mnew) : 0.0f;
            const float p1 = b1 ? __expf(s1[r] - mnew) : 0.0f;
            psum += p0 + p1;
            bp[r]     = (_Float16)p0;
            bp[r + 8] = (_Float16)p1;
        }
        psum += __shfl_xor(psum, 16, 32);
        lrow = lrow * alpha + psum;
        mrow = mnew;

#pragma unroll
        for (int j = 0; j < 8; ++j) {
            o0[j] *= alpha; o1[j] *= alpha; o2[j] *= alpha; o3[j] *= alpha;
        }

        // ---- O^T += V^T . P^T  (4 dk-tiles), A from LDS
#pragma unroll
        for (int t = 0; t < 4; ++t) {
            const _Float16* vr = &sV[cur][(t * 16 + lm) * 32];
            const v8h lo = *(const v8h*)(vr + koff);
            const v8h hi = *(const v8h*)(vr + 16 + koff);
            v16h a;
#pragma unroll
            for (int i = 0; i < 8; ++i) { a[i] = lo[i]; a[i + 8] = hi[i]; }
            v8f* ot = (t == 0) ? &o0 : (t == 1) ? &o1 : (t == 2) ? &o2 : &o3;
            *ot = WMMA_F16(a, bp, *ot);
        }

        __syncthreads(); // all waves done reading buf[cur] before reuse
    }

    // ---- normalize + store: O^T element (dk = t*16 + r + half*8, q = q0+lm)
    const float inv = 1.0f / lrow;
    _Float16* orow = of + ((size_t)b * kS + q) * kDM + h * kDK;
#pragma unroll
    for (int t = 0; t < 4; ++t) {
        const v8f ot = (t == 0) ? o0 : (t == 1) ? o1 : (t == 2) ? o2 : o3;
#pragma unroll
        for (int r = 0; r < 8; ++r) {
            const int dk = t * 16 + r + half * 8;
            orow[dk] = (_Float16)(ot[r] * inv);
        }
    }
}

// ---------------------------------------------------------------------------
// Kernel 3: output projection. out = attn(f16) @ Wo^T + bo, f32 store.
// ---------------------------------------------------------------------------
__global__ __launch_bounds__(32) void k_out_proj(
    const _Float16* __restrict__ of, const float* __restrict__ wo,
    const float* __restrict__ bo, float* __restrict__ out)
{
    const int lane = threadIdx.x & 31;
    const int half = lane >> 4;
    const int lm   = lane & 15;
    const int koff = half ? 8 : 0;

    const int m0 = blockIdx.x * 16;
    const int n0 = blockIdx.y * 64;

    v8f acc[4] = {};
    const _Float16* ar = of + (size_t)(m0 + lm) * kDM;
    for (int k0 = 0; k0 < kDM; k0 += 32) {
        const v8h lo = *(const v8h*)(ar + k0 + koff);
        const v8h hi = *(const v8h*)(ar + k0 + 16 + koff);
        v16h a;
#pragma unroll
        for (int i = 0; i < 8; ++i) { a[i] = lo[i]; a[i + 8] = hi[i]; }
#pragma unroll
        for (int t = 0; t < 4; ++t) {
            const float* wr = wo + (size_t)(n0 + t * 16 + lm) * kDM + k0;
            v16h bm;
#pragma unroll
            for (int i = 0; i < 8; ++i) {
                bm[i]     = (_Float16)wr[koff + i];
                bm[i + 8] = (_Float16)wr[16 + koff + i];
            }
            acc[t] = WMMA_F16(a, bm, acc[t]);
        }
    }

#pragma unroll
    for (int t = 0; t < 4; ++t) {
        const int n = n0 + t * 16 + lm;
        const float bval = bo[n];
#pragma unroll
        for (int r = 0; r < 8; ++r) {
            const int m = m0 + r + half * 8;
            out[(size_t)m * kDM + n] = acc[t][r] + bval;
        }
    }
}

// ---------------------------------------------------------------------------
extern "C" void kernel_launch(void* const* d_in, const int* in_sizes, int n_in,
                              void* d_out, int out_size, void* d_ws,
                              size_t ws_size, hipStream_t stream) {
    (void)in_sizes; (void)n_in; (void)out_size; (void)ws_size;

    const float* x    = (const float*)d_in[0];
    const int*   dmsk = (const int*)d_in[1];
    const float* wq   = (const float*)d_in[2];
    const float* bq   = (const float*)d_in[3];
    const float* wk   = (const float*)d_in[4];
    const float* bk   = (const float*)d_in[5];
    const float* wv   = (const float*)d_in[6];
    const float* bv   = (const float*)d_in[7];
    const float* wo   = (const float*)d_in[8];
    const float* bo   = (const float*)d_in[9];
    float* out = (float*)d_out;

    const size_t elems = (size_t)kB * kH * kS * kDK; // 4,194,304 per tensor
    _Float16* qf = (_Float16*)d_ws;
    _Float16* kf = qf + elems;
    _Float16* vf = kf + elems;
    _Float16* of = vf + elems;
    unsigned* mbits = (unsigned*)(of + elems);       // +2MB packed mask

    const int nmaskwords = (kS * kS) / 32; // 524288
    k_mask_pack<<<nmaskwords / 256, 256, 0, stream>>>(dmsk, mbits);

    dim3 g1((kB * kS) / 16, kH);
    k_proj_qkv<<<g1, 32, 0, stream>>>(x, wq, bq, wk, bk, wv, bv, qf, kf, vf);

    const int nblk = kB * kH * 32; // (b,h) x 32 q-groups = 512 blocks
    k_attn<<<nblk, 256, 0, stream>>>(qf, kf, vf, mbits, of);

    dim3 g3((kB * kS) / 16, kDM / 64);
    k_out_proj<<<g3, 32, 0, stream>>>(of, wo, bo, out);
}
